// Transformer_47158740910429
// MI455X (gfx1250) — matrix operand order repaired and checked by reading, not compiled, not run on hardware
//
#include <hip/hip_runtime.h>
#include <hip/hip_bf16.h>
#include <stdint.h>

// ---------------------------------------------------------------------------
// Types for CDNA5 WMMA (gfx1250, wave32)
// ---------------------------------------------------------------------------
typedef __attribute__((ext_vector_type(16))) __bf16 v16bf;
typedef __attribute__((ext_vector_type(8)))  float  v8f;

#define BLK_M 128
#define BLK_N 64
#define BLK_K 32
#define LDSTR 40    // bf16 elems per row of WMMA tile: 80B stride -> 16B-aligned frags
#define A32STR 36   // f32 elems per row of raw A staging tile (144B, 16B-aligned)
#define BT32STR 36  // raw B staging row stride, btrans mode ([n][k], 64 rows)
#define BN32STR 68  // raw B staging row stride, normal mode ([k][n], 32 rows; 272B)

__device__ __forceinline__ unsigned short f2bf(float f) {
  unsigned int u = __float_as_uint(f);
  u += 0x7FFFu + ((u >> 16) & 1u);   // round-to-nearest-even
  return (unsigned short)(u >> 16);
}
__device__ __forceinline__ unsigned int pack2bf(float a, float b) {
  return (unsigned int)f2bf(a) | ((unsigned int)f2bf(b) << 16);
}

union FragU {
  uint4 q[2];
  v16bf v;
};

// CDNA5 async DMA: global -> LDS, 16B per lane, tracked by ASYNCcnt
__device__ __forceinline__ void async_b128(unsigned lds_off, const float* gaddr) {
  asm volatile("global_load_async_to_lds_b128 %0, %1, off"
               :: "v"(lds_off), "v"(gaddr) : "memory");
}
__device__ __forceinline__ void wait_async0() {
  asm volatile("s_wait_asynccnt 0x0" ::: "memory");
}

// ---------------------------------------------------------------------------
// Batched-strided GEMM: C[M,N] = op(relu)( A[M,K] * B[K,N] ), bf16 WMMA, f32 acc
//   A: f32 row-major (lda), DMA'd to LDS then converted to bf16
//   B: f32; btrans==0 -> B[k*ldb+n]; btrans==1 -> B[n*ldb+k] (C = A*B^T view)
//   batch z decomposed as (zb = z/Hdiv, zh = z%Hdiv) with independent strides
// grid: (M/128, N/64, batches), block: 256 threads (8 wave32)
// Requires: lda/ldb/ldc multiples of 4; rowBase/colBase/k0 tile-aligned (all
// dims in this model are multiples of 128/64/32, so no predication -> EXEC
// stays all-ones as WMMA requires).
// ---------------------------------------------------------------------------
__global__ __launch_bounds__(256) void gemm_bf16_wmma(
    const float* __restrict__ A, const float* __restrict__ B, float* __restrict__ C,
    int lda, int ldb, int ldc, int Ktot, int btrans, int relu, int Hdiv,
    long sa_b, long sa_h, long sb_b, long sb_h, long sc_b, long sc_h)
{
  __shared__ __align__(16) float          As32[BLK_M * A32STR];        // raw fp32 A tile
  __shared__ __align__(16) float          Bs32[BLK_N * BT32STR];       // raw fp32 B tile (both modes fit)
  __shared__ __align__(16) unsigned short As[BLK_M * LDSTR];           // bf16 [m][k]
  __shared__ __align__(16) unsigned short Bs[BLK_N * LDSTR];           // bf16 [n][k]

  const int zb = blockIdx.z / Hdiv;
  const int zh = blockIdx.z % Hdiv;
  A += (long)zb * sa_b + (long)zh * sa_h;
  B += (long)zb * sb_b + (long)zh * sb_h;
  C += (long)zb * sc_b + (long)zh * sc_h;

  const int tid  = threadIdx.x;
  const int lane = tid & 31;
  const int wave = tid >> 5;
  const int wm   = wave >> 1;   // 0..3 : 32-row strip
  const int wn   = wave & 1;    // 0..1 : 32-col strip
  const int rowBase = blockIdx.x * BLK_M;
  const int colBase = blockIdx.y * BLK_N;

  const unsigned offA32 = (unsigned)(uintptr_t)As32;  // LDS aperture: offset in addr[31:0]
  const unsigned offB32 = (unsigned)(uintptr_t)Bs32;

  v8f acc[2][2] = {};

  for (int k0 = 0; k0 < Ktot; k0 += BLK_K) {
    // ---- phase 1: async DMA raw fp32 tiles into LDS (ASYNCcnt) ----
    // A: 128 rows x 8 x 16B granules
    for (int g = tid; g < BLK_M * 8; g += 256) {
      int r = g >> 3, c = (g & 7) * 4;
      async_b128(offA32 + (unsigned)(r * A32STR + c) * 4,
                 &A[(long)(rowBase + r) * lda + (k0 + c)]);
    }
    if (btrans) {   // B[n][k]: 64 rows x 8 granules
      for (int g = tid; g < BLK_N * 8; g += 256) {
        int n = g >> 3, c = (g & 7) * 4;
        async_b128(offB32 + (unsigned)(n * BT32STR + c) * 4,
                   &B[(long)(colBase + n) * ldb + (k0 + c)]);
      }
    } else {        // B[k][n]: 32 rows x 16 granules of 4 along n
      for (int g = tid; g < BLK_K * 16; g += 256) {
        int kk = g >> 4, n = (g & 15) * 4;
        async_b128(offB32 + (unsigned)(kk * BN32STR + n) * 4,
                   &B[(long)(k0 + kk) * ldb + (colBase + n)]);
      }
    }
    wait_async0();
    __syncthreads();

    // ---- phase 2: packed f32 -> bf16 conversion into WMMA-layout tiles ----
    for (int g = tid; g < BLK_M * 8; g += 256) {
      int r = g >> 3, c = (g & 7) * 4;
      float4 v = *(const float4*)&As32[r * A32STR + c];
      uint2 p; p.x = pack2bf(v.x, v.y); p.y = pack2bf(v.z, v.w);
      *(uint2*)&As[r * LDSTR + c] = p;
    }
    if (btrans) {
      for (int g = tid; g < BLK_N * 8; g += 256) {
        int n = g >> 3, c = (g & 7) * 4;
        float4 v = *(const float4*)&Bs32[n * BT32STR + c];
        uint2 p; p.x = pack2bf(v.x, v.y); p.y = pack2bf(v.z, v.w);
        *(uint2*)&Bs[n * LDSTR + c] = p;
      }
    } else {
      for (int g = tid; g < BLK_K * 16; g += 256) {
        int kk = g >> 4, n = (g & 15) * 4;
        float4 v = *(const float4*)&Bs32[kk * BN32STR + n];
        Bs[(n + 0) * LDSTR + kk] = f2bf(v.x);
        Bs[(n + 1) * LDSTR + kk] = f2bf(v.y);
        Bs[(n + 2) * LDSTR + kk] = f2bf(v.z);
        Bs[(n + 3) * LDSTR + kk] = f2bf(v.w);
      }
    }
    __syncthreads();

    // ---- phase 3: fragment loads (ISA wave32 16-bit A/B layouts) + WMMA ----
    v16bf af[2], bfv[2];
#pragma unroll
    for (int i = 0; i < 2; ++i) {
      int m  = wm * 32 + i * 16 + (lane & 15);
      int ko = (lane >> 4) * 8;                       // lanes 16-31 hold K+8 group
      FragU f;
      f.q[0] = *(const uint4*)&As[m * LDSTR + ko];        // K = ko .. ko+7
      f.q[1] = *(const uint4*)&As[m * LDSTR + ko + 16];   // K = 16+ko .. 16+ko+7
      af[i] = f.v;
    }
#pragma unroll
    for (int j = 0; j < 2; ++j) {
      int n  = wn * 32 + j * 16 + (lane & 15);
      int ko = (lane >> 4) * 8;                       // lanes 16-31 hold K+8 group
      const uint4* p = (const uint4*)&Bs[n * LDSTR + ko];
      FragU f;
      f.q[0] = p[0];
      f.q[1] = p[2];
      bfv[j] = f.v;
    }
#pragma unroll
    for (int i = 0; i < 2; ++i)
#pragma unroll
      for (int j = 0; j < 2; ++j)
        acc[i][j] = __builtin_amdgcn_wmma_f32_16x16x32_bf16(
            false, af[i], false, bfv[j], (short)0, acc[i][j], false, false);
    __syncthreads();
  }

  // C/D layout: VGPR r holds M = tileRow + r + 8*(lane>=16), N = tileCol + (lane&15)
#pragma unroll
  for (int i = 0; i < 2; ++i)
#pragma unroll
    for (int j = 0; j < 2; ++j) {
      int m0 = rowBase + wm * 32 + i * 16 + (lane >> 4) * 8;
      int n  = colBase + wn * 32 + j * 16 + (lane & 15);
#pragma unroll
      for (int r = 0; r < 8; ++r) {
        float v = acc[i][j][r];
        if (relu) v = fmaxf(v, 0.0f);
        C[(long)(m0 + r) * ldc + n] = v;
      }
    }
}

// ---------------------------------------------------------------------------
// Row softmax over 512 elems (scores laid out [B,H,S,S]); scale + causal mask
// ---------------------------------------------------------------------------
__global__ __launch_bounds__(256) void softmax512(float* __restrict__ S, int causal, float scale)
{
  __shared__ float red[256];
  const long row = blockIdx.x;
  const int  q   = (int)(row & 511);          // query index within [S,S] block
  float* p = S + row * 512;
  const int tid = threadIdx.x;

  float v0 = p[tid] * scale;
  float v1 = p[tid + 256] * scale;
  if (causal) {
    if (tid > q)       v0 = -1e9f;
    if (tid + 256 > q) v1 = -1e9f;
  }
  red[tid] = fmaxf(v0, v1);
  __syncthreads();
  for (int s = 128; s > 0; s >>= 1) { if (tid < s) red[tid] = fmaxf(red[tid], red[tid + s]); __syncthreads(); }
  float mx = red[0];
  __syncthreads();
  float e0 = __expf(v0 - mx), e1 = __expf(v1 - mx);
  red[tid] = e0 + e1;
  __syncthreads();
  for (int s = 128; s > 0; s >>= 1) { if (tid < s) red[tid] += red[tid + s]; __syncthreads(); }
  float inv = 1.0f / red[0];
  p[tid]       = e0 * inv;
  p[tid + 256] = e1 * inv;
}

// ---------------------------------------------------------------------------
// R = LayerNorm(X + R) over rows of 512 (weight=1, bias=0, eps=1e-5)
// ---------------------------------------------------------------------------
__global__ __launch_bounds__(256) void add_ln512(const float* __restrict__ X, float* __restrict__ R)
{
  __shared__ float red[256];
  const long row = blockIdx.x;
  const float* px = X + row * 512;
  float* pr = R + row * 512;
  const int tid = threadIdx.x;

  float a0 = px[tid] + pr[tid];
  float a1 = px[tid + 256] + pr[tid + 256];
  red[tid] = a0 + a1;
  __syncthreads();
  for (int s = 128; s > 0; s >>= 1) { if (tid < s) red[tid] += red[tid + s]; __syncthreads(); }
  float mean = red[0] * (1.0f / 512.0f);
  __syncthreads();
  float d0 = a0 - mean, d1 = a1 - mean;
  red[tid] = d0 * d0 + d1 * d1;
  __syncthreads();
  for (int s = 128; s > 0; s >>= 1) { if (tid < s) red[tid] += red[tid + s]; __syncthreads(); }
  float rstd = rsqrtf(red[0] * (1.0f / 512.0f) + 1e-5f);
  pr[tid]       = d0 * rstd;
  pr[tid + 256] = d1 * rstd;
}

// ---------------------------------------------------------------------------
// Host-side sequencing
// ---------------------------------------------------------------------------
static inline void launch_gemm(hipStream_t st, const float* A, const float* B, float* C,
    int M, int N, int K, int lda, int ldb, int ldc, int btrans, int relu,
    int batches, int Hdiv, long sab, long sah, long sbb, long sbh, long scb, long sch)
{
  dim3 grid(M / BLK_M, N / BLK_N, batches);
  gemm_bf16_wmma<<<grid, 256, 0, st>>>(A, B, C, lda, ldb, ldc, K, btrans, relu, Hdiv,
                                       sab, sah, sbb, sbh, scb, sch);
}

// Multi-head attention sublayer: xq <- LN( MHA(xq, xkv) + xq )
static void attention(hipStream_t st, float* xq, const float* xkv,
                      const float* Wq, const float* Wk, const float* Wv, const float* Wo,
                      int causal, float* Qb, float* Kb, float* Vb, float* Sc, float* Ctx, float* Tmp)
{
  const int M = 2048;                       // B*S rows
  const long SD = 512L * 512;               // per-batch activation stride [S, D]
  // Q/K/V projections (M x 512) = (M x 512) @ (512 x 512)
  launch_gemm(st, xq,  Wq, Qb, M, 512, 512, 512, 512, 512, 0, 0, 1, 1, 0,0,0,0,0,0);
  launch_gemm(st, xkv, Wk, Kb, M, 512, 512, 512, 512, 512, 0, 0, 1, 1, 0,0,0,0,0,0);
  launch_gemm(st, xkv, Wv, Vb, M, 512, 512, 512, 512, 512, 0, 0, 1, 1, 0,0,0,0,0,0);
  // scores[b,h] = Q[b,:,h] @ K[b,:,h]^T   (512x512, K=64), batch = B*H = 32
  launch_gemm(st, Qb, Kb, Sc, 512, 512, 64, 512, 512, 512, /*btrans=*/1, 0,
              32, 8, SD, 64, SD, 64, 8L * SD, SD);
  softmax512<<<dim3(32 * 512), 256, 0, st>>>(Sc, causal, 0.125f);   // 1/sqrt(64)
  // ctx[b,h] = attn[b,h] @ V[b,:,h]       (512x64, K=512)
  launch_gemm(st, Sc, Vb, Ctx, 512, 64, 512, 512, 512, 512, 0, 0,
              32, 8, 8L * SD, SD, SD, 64, SD, 64);
  // out = ctx @ Wo  (heads already concatenated in Ctx's [B,S,H*Dv] layout)
  launch_gemm(st, Ctx, Wo, Tmp, M, 512, 512, 512, 512, 512, 0, 0, 1, 1, 0,0,0,0,0,0);
  add_ln512<<<dim3(2048), 256, 0, st>>>(Tmp, xq);
}

// FFN sublayer: h <- LN( relu(h@W1)@W2 + h )
static void ffn_block(hipStream_t st, float* h, const float* W1, const float* W2,
                      float* F1, float* Tmp)
{
  launch_gemm(st, h,  W1, F1,  2048, 2048, 512,  512, 2048, 2048, 0, /*relu=*/1, 1, 1, 0,0,0,0,0,0);
  launch_gemm(st, F1, W2, Tmp, 2048, 512,  2048, 2048, 512,  512,  0, 0,         1, 1, 0,0,0,0,0,0);
  add_ln512<<<dim3(2048), 256, 0, st>>>(Tmp, h);
}

extern "C" void kernel_launch(void* const* d_in, const int* in_sizes, int n_in,
                              void* d_out, int out_size, void* d_ws, size_t ws_size,
                              hipStream_t stream)
{
  (void)in_sizes; (void)n_in; (void)out_size; (void)ws_size;
  const float* enc_in  = (const float*)d_in[0];
  const float* dec_in  = (const float*)d_in[1];
  // d_in[2] = dec_self_attn_mask (exact lower-triangular ones) -> causal flag
  const float* enc_Wq  = (const float*)d_in[3];
  const float* enc_Wk  = (const float*)d_in[4];
  const float* enc_Wv  = (const float*)d_in[5];
  const float* enc_Wo  = (const float*)d_in[6];
  const float* enc_W1  = (const float*)d_in[7];
  const float* enc_W2  = (const float*)d_in[8];
  const float* dsa_Wq  = (const float*)d_in[9];
  const float* dsa_Wk  = (const float*)d_in[10];
  const float* dsa_Wv  = (const float*)d_in[11];
  const float* dsa_Wo  = (const float*)d_in[12];
  const float* dca_Wq  = (const float*)d_in[13];
  const float* dca_Wk  = (const float*)d_in[14];
  const float* dca_Wv  = (const float*)d_in[15];
  const float* dca_Wo  = (const float*)d_in[16];
  const float* dec_W1  = (const float*)d_in[17];
  const float* dec_W2  = (const float*)d_in[18];

  // Workspace carve-up (floats)
  float* ws = (float*)d_ws;
  const long ACT = 2048L * 512;
  float* Hc  = ws;            // current activations [B*S, D]
  float* Enc = Hc  + ACT;     // encoder output
  float* Qb  = Enc + ACT;
  float* Kb  = Qb  + ACT;
  float* Vb  = Kb  + ACT;
  float* Ctx = Vb  + ACT;
  float* Tmp = Ctx + ACT;     // sublayer output before residual+LN
  float* F1  = Tmp + ACT;     // FFN hidden [2048, 2048]
  float* Sc  = F1  + 2048L * 2048;  // scores/attn [B,H,S,S] f32 (32 MB)

  const long WA = 512L * 512;   // per-layer attention weight stride
  const long WF = 512L * 2048;  // per-layer FFN weight stride

  // ---- Encoder ----
  hipMemcpyAsync(Hc, enc_in, ACT * sizeof(float), hipMemcpyDeviceToDevice, stream);
  for (int l = 0; l < 6; ++l) {
    attention(stream, Hc, Hc, enc_Wq + l * WA, enc_Wk + l * WA, enc_Wv + l * WA,
              enc_Wo + l * WA, /*causal=*/0, Qb, Kb, Vb, Sc, Ctx, Tmp);
    ffn_block(stream, Hc, enc_W1 + l * WF, enc_W2 + l * WF, F1, Tmp);
  }
  hipMemcpyAsync(Enc, Hc, ACT * sizeof(float), hipMemcpyDeviceToDevice, stream);

  // ---- Decoder ----
  hipMemcpyAsync(Hc, dec_in, ACT * sizeof(float), hipMemcpyDeviceToDevice, stream);
  for (int l = 0; l < 6; ++l) {
    attention(stream, Hc, Hc, dsa_Wq + l * WA, dsa_Wk + l * WA, dsa_Wv + l * WA,
              dsa_Wo + l * WA, /*causal=*/1, Qb, Kb, Vb, Sc, Ctx, Tmp);
    attention(stream, Hc, Enc, dca_Wq + l * WA, dca_Wk + l * WA, dca_Wv + l * WA,
              dca_Wo + l * WA, /*causal=*/0, Qb, Kb, Vb, Sc, Ctx, Tmp);
    ffn_block(stream, Hc, dec_W1 + l * WF, dec_W2 + l * WF, F1, Tmp);
  }

  hipMemcpyAsync(d_out, Hc, ACT * sizeof(float), hipMemcpyDeviceToDevice, stream);
}
